// SceneGraphNet_49615462203562
// MI455X (gfx1250) — matrix operand order repaired and checked by reading, not compiled, not run on hardware
//
#include <hip/hip_runtime.h>

typedef __attribute__((ext_vector_type(16))) _Float16 v16h;
typedef __attribute__((ext_vector_type(8)))  _Float16 v8h;
typedef __attribute__((ext_vector_type(8)))  float    v8f;

#define MM   16384   // clusters
#define NPTS 131072  // points
#define DD   64      // feature dim

// ---------------------------------------------------------------- utilities

__global__ void k_fill_u32(unsigned int* p, unsigned int v, int n) {
  int i = blockIdx.x * blockDim.x + threadIdx.x;
  if (i < n) p[i] = v;
}

// f32 weight (K x N row major) -> f16 swizzled into WMMA B-fragment order:
// dst[((ntile*(Kp/32) + kc)*32 + lane)*16 + p],
//   k = kc*32 + (p/8)*16 + (lane/16)*8 + p%8,  n = ntile*16 + lane%16
__global__ void k_cvt_weight_sw(const float* __restrict__ src, _Float16* __restrict__ dst,
                                int K, int N, int Kp, int Np) {
  int t = blockIdx.x * blockDim.x + threadIdx.x;
  int total = (Np >> 4) * (Kp >> 5) * 512;
  if (t >= total) return;
  int p    = t & 15;
  int lane = (t >> 4) & 31;
  int rest = t >> 9;                 // ntile*(Kp/32) + kc
  int nkc  = Kp >> 5;
  int ntile = rest / nkc, kc = rest - ntile * nkc;
  int k = (kc << 5) + ((p >> 3) << 4) + ((lane >> 4) << 3) + (p & 7);
  int n = (ntile << 4) + (lane & 15);
  float v = (k < K && n < N) ? src[k * N + n] : 0.0f;
  dst[t] = (_Float16)v;
}

static __device__ __forceinline__ void atomicMaxF32(float* addr, float val) {
  // ordered-int trick; buffer initialized to -inf bit pattern (0xFF800000)
  if (val >= 0.0f) atomicMax((int*)addr, __float_as_int(val));
  else             atomicMin((unsigned int*)addr, __float_as_uint(val));
}

// ---------------------------------------------------------------- segment sums

__global__ void k_scatter(const float* __restrict__ x, const float* __restrict__ coords,
                          const int* __restrict__ cluster,
                          float* cnt, float* sumx, float* sumc, float* summ) {
  long long t = (long long)blockIdx.x * blockDim.x + threadIdx.x;
  if (t >= (long long)NPTS * DD) return;
  int i = (int)(t >> 6), d = (int)(t & 63);
  int c = cluster[i];
  atomicAdd(sumx + (size_t)c * DD + d, x[t]);
  if (d == 0) atomicAdd(cnt + c, 1.0f);
  if (d < 2) {
    float cv = coords[2 * i + d];
    atomicAdd(sumc + 2 * c + d, cv);
    atomicAdd(summ + 2 * c + d, cv * cv);
  }
}

// h0 = [xp, 10*cent, 10*mom] padded to 128 cols, stored f16; also cent (f32)
__global__ void k_build_h0(const float* __restrict__ cnt, const float* __restrict__ sumx,
                           const float* __restrict__ sumc, const float* __restrict__ summ,
                           _Float16* __restrict__ actA, float* __restrict__ cent) {
  int t = blockIdx.x * blockDim.x + threadIdx.x;
  if (t >= MM * 128) return;
  int m = t >> 7, col = t & 127;
  float inv = 1.0f / fmaxf(cnt[m], 1.0f);
  float v = 0.0f;
  if (col < 64) {
    v = sumx[(size_t)m * 64 + col] * inv;
  } else if (col < 66) {
    float ce = sumc[2 * m + (col - 64)] * inv;
    cent[2 * m + (col - 64)] = ce;
    v = 10.0f * ce;
  } else if (col < 68) {
    v = 10.0f * summ[2 * m + (col - 66)] * inv;
  }
  actA[t] = (_Float16)v;
}

// ---------------------------------------------------------------- WMMA GEMM
// One wave computes one 16x16 tile of  D = act(A[M x KLOOP] @ W[KLOOP x Np] + b)
// A (row major f16, stride lda): lane holds row m=lane%16; per 32-K chunk two
//   contiguous 8-half runs at k0+khalf*8 and k0+16+khalf*8  -> 2x b128 loads.
// W pre-swizzled to fragment order -> 32 contiguous bytes per lane -> 2x b128.
// C/D: row = r + 8*(lane/16), col = lane%16.
template <int KLOOP, bool HASH, bool HASF>
__global__ void k_gemm_wmma(const _Float16* __restrict__ A, int lda,
                            const _Float16* __restrict__ Wsw,
                            const float* __restrict__ bias, int Nvalid, int relu,
                            _Float16* __restrict__ outH, int ldh,
                            float* __restrict__ outF, int ldf) {
  constexpr int NKC = KLOOP / 32;
  int lane  = threadIdx.x;             // 0..31, wave32, EXEC all ones
  int ncol  = lane & 15;
  int khalf = lane >> 4;
  int gm = blockIdx.x * 16;
  int gn16 = blockIdx.y;               // n-tile index
  const _Float16* Arow = A + (size_t)(gm + ncol) * lda + (khalf << 3);
  const _Float16* Wl   = Wsw + (((size_t)gn16 * NKC) * 32 + lane) * 16;
  v8f acc = {};
  #pragma unroll
  for (int kc = 0; kc < NKC; ++kc) {
    __builtin_prefetch(Arow + kc * 32 + 32, 0, 0);        // global_prefetch_b8
    v8h alo = *(const v8h*)(Arow + kc * 32);
    v8h ahi = *(const v8h*)(Arow + kc * 32 + 16);
    v8h blo = *(const v8h*)(Wl + (size_t)kc * 512);
    v8h bhi = *(const v8h*)(Wl + (size_t)kc * 512 + 8);
    v16h a = __builtin_shufflevector(alo, ahi, 0,1,2,3,4,5,6,7,8,9,10,11,12,13,14,15);
    v16h b = __builtin_shufflevector(blo, bhi, 0,1,2,3,4,5,6,7,8,9,10,11,12,13,14,15);
    acc = __builtin_amdgcn_wmma_f32_16x16x32_f16(
        /*neg_a=*/false, a, /*neg_b=*/false, b,
        /*c_mod=*/(short)0, acc, /*reuse_a=*/false, /*reuse_b=*/false);
  }
  int n  = gn16 * 16 + ncol;
  int m0 = gm + (khalf << 3);
  float bn = (n < Nvalid) ? bias[n] : 0.0f;     // Nvalid==0 => never dereferenced
  float res[8];
  #pragma unroll
  for (int r = 0; r < 8; ++r) {
    float v = acc[r] + bn;
    res[r] = relu ? fmaxf(v, 0.0f) : v;
  }
  if constexpr (HASH) {
    _Float16* ph = outH + (size_t)m0 * ldh + n;
    #pragma unroll
    for (int r = 0; r < 8; ++r) { *ph = (_Float16)res[r]; ph += ldh; }
  }
  if constexpr (HASF) {
    float* pf = outF + (size_t)m0 * ldf + n;
    #pragma unroll
    for (int r = 0; r < 8; ++r) { *pf = res[r]; pf += ldf; }
  }
}

// ---------------------------------------------------------------- edge segment-max

__global__ void k_edge_max(const int* __restrict__ esrc, const int* __restrict__ edst,
                           const int* __restrict__ cluster, const float* __restrict__ hWn,
                           float* agg, int E) {
  long long t = (long long)blockIdx.x * blockDim.x + threadIdx.x;
  if (t >= (long long)E * DD) return;
  int e = (int)(t >> 6), d = (int)(t & 63);
  int cs = cluster[esrc[e]];
  int cd = cluster[edst[e]];
  atomicMaxF32(agg + (size_t)cd * DD + d, hWn[(size_t)cs * DD + d]);
}

// feat = h@Wr + (finite(agg)?agg:0) + bg  -> f16 act buffer (cols 0..63)
__global__ void k_feat(const float* __restrict__ hWr, const float* __restrict__ agg,
                       const float* __restrict__ bg, _Float16* __restrict__ actA) {
  int t = blockIdx.x * blockDim.x + threadIdx.x;
  if (t >= MM * DD) return;
  int m = t >> 6, d = t & 63;
  float a = agg[t];
  if (!__builtin_isfinite(a)) a = 0.0f;
  actA[(size_t)m * 128 + d] = (_Float16)(hWr[t] + a + bg[d]);
}

// ---------------------------------------------------------------- render

__global__ void k_render(const float* __restrict__ coords, const int* __restrict__ cluster,
                         const float* __restrict__ qout, const float* __restrict__ cent,
                         float* __restrict__ out) {
  int i = blockIdx.x * blockDim.x + threadIdx.x;
  if (i >= NPTS) return;
  int c = cluster[i];
  float jsf[20];
  #pragma unroll
  for (int j = 0; j < 18; ++j) jsf[j] = qout[(size_t)c * 32 + j];
  jsf[18] = cent[2 * c];
  jsf[19] = cent[2 * c + 1];
  float dx = coords[2 * i]     - jsf[0];   // centers = feats[:, :2] (q's first 2 cols)
  float dy = coords[2 * i + 1] - jsf[1];
  #pragma unroll
  for (int k = 0; k < 3; ++k) {
    const float* p = jsf + 2 + 6 * k;      // paras[k][0..5]
    out[(size_t)i * 3 + k] =
        p[0] + p[1] * dx + p[2] * dy + p[3] * dx * dx + p[4] * dy * dy + p[5] * dx * dy;
  }
}

// ---------------------------------------------------------------- launch

extern "C" void kernel_launch(void* const* d_in, const int* in_sizes, int n_in,
                              void* d_out, int out_size, void* d_ws, size_t ws_size,
                              hipStream_t stream) {
  const float* x      = (const float*)d_in[0];
  const float* coords = (const float*)d_in[1];
  const int*   cluster= (const int*)  d_in[2];
  const int*   esrc   = (const int*)  d_in[3];
  const int*   edst   = (const int*)  d_in[4];
  const float* W1 = (const float*)d_in[5];  const float* b1 = (const float*)d_in[6];
  const float* W2 = (const float*)d_in[7];  const float* b2 = (const float*)d_in[8];
  const float* W3 = (const float*)d_in[9];  const float* b3 = (const float*)d_in[10];
  const float* Wr = (const float*)d_in[11]; const float* Wn = (const float*)d_in[12];
  const float* bg = (const float*)d_in[13];
  const float* Q1w = (const float*)d_in[14]; const float* Q1b = (const float*)d_in[15];
  const float* Q2w = (const float*)d_in[16]; const float* Q2b = (const float*)d_in[17];
  const float* Q3w = (const float*)d_in[18]; const float* Q3b = (const float*)d_in[19];
  const float* Q4w = (const float*)d_in[20]; const float* Q4b = (const float*)d_in[21];
  const int E = in_sizes[3];
  (void)n_in; (void)out_size; (void)ws_size;

  char* ws = (char*)d_ws;
  size_t off = 0;
  auto carve = [&](size_t bytes) -> char* {
    char* p = ws + off; off += (bytes + 255) & ~(size_t)255; return p;
  };
  float*    cnt  = (float*)carve((size_t)MM * 4);
  float*    sumx = (float*)carve((size_t)MM * 64 * 4);
  float*    sumc = (float*)carve((size_t)MM * 2 * 4);
  float*    summ = (float*)carve((size_t)MM * 2 * 4);
  float*    cent = (float*)carve((size_t)MM * 2 * 4);
  float*    hWn  = (float*)carve((size_t)MM * 64 * 4);
  float*    hWr  = (float*)carve((size_t)MM * 64 * 4);
  float*    agg  = (float*)carve((size_t)MM * 64 * 4);
  float*    qout = (float*)carve((size_t)MM * 32 * 4);
  _Float16* actA = (_Float16*)carve((size_t)MM * 128 * 2);
  _Float16* actB = (_Float16*)carve((size_t)MM * 128 * 2);
  _Float16* wp[9];
  for (int i = 0; i < 9; ++i) wp[i] = (_Float16*)carve((size_t)128 * 112 * 2);

  const int TB = 256;
  auto nb = [](long long n, int tb) { return dim3((unsigned)((n + tb - 1) / tb)); };

  // init: zero accumulators + actB (its 112..127 pad columns must be 0), agg = -inf bits
  k_fill_u32<<<nb(MM, TB), TB, 0, stream>>>((unsigned*)cnt,  0u, MM);
  k_fill_u32<<<nb((long long)MM * 64, TB), TB, 0, stream>>>((unsigned*)sumx, 0u, MM * 64);
  k_fill_u32<<<nb(MM * 2, TB), TB, 0, stream>>>((unsigned*)sumc, 0u, MM * 2);
  k_fill_u32<<<nb(MM * 2, TB), TB, 0, stream>>>((unsigned*)summ, 0u, MM * 2);
  k_fill_u32<<<nb((long long)MM * 64, TB), TB, 0, stream>>>((unsigned*)agg, 0xFF800000u, MM * 64);
  k_fill_u32<<<nb((long long)MM * 64, TB), TB, 0, stream>>>((unsigned*)actB, 0u, MM * 64);

  // weight conversion to fragment-swizzled f16 tiles
  struct CV { const float* src; int K, N, Kp, Np; } cv[9] = {
    {W1,  68, 100,  96, 112}, {W2, 100, 100, 128, 112}, {W3, 100, 64, 128, 64},
    {Wr,  64,  64,  64,  64}, {Wn,  64,  64,  64,  64},
    {Q1w, 64, 100,  64, 112}, {Q2w,100, 100, 128, 112}, {Q3w,100,100, 128, 112},
    {Q4w,100,  18, 128,  32}};
  for (int i = 0; i < 9; ++i)
    k_cvt_weight_sw<<<nb((long long)cv[i].Kp * cv[i].Np, TB), TB, 0, stream>>>(
        cv[i].src, wp[i], cv[i].K, cv[i].N, cv[i].Kp, cv[i].Np);

  // segment sums and h0
  k_scatter<<<nb((long long)NPTS * DD, TB), TB, 0, stream>>>(x, coords, cluster,
                                                             cnt, sumx, sumc, summ);
  k_build_h0<<<nb((long long)MM * 128, TB), TB, 0, stream>>>(cnt, sumx, sumc, summ, actA, cent);

  dim3 wv(32);
  auto G = [](int Np) { return dim3(MM / 16, (unsigned)(Np / 16)); };

  // encoder MLP
  k_gemm_wmma< 96, true, false><<<G(112), wv, 0, stream>>>(
      actA, 128, wp[0], b1, 100, 1, actB, 128, nullptr, 0);
  k_gemm_wmma<128, true, false><<<G(112), wv, 0, stream>>>(
      actB, 128, wp[1], b2, 100, 1, actA, 128, nullptr, 0);
  k_gemm_wmma<128, true, false><<<G(64), wv, 0, stream>>>(
      actA, 128, wp[2], b3, 64, 0, actB, 128, nullptr, 0);
  // h @ Wn, h @ Wr   (no bias: Nvalid = 0)
  k_gemm_wmma< 64, false, true><<<G(64), wv, 0, stream>>>(
      actB, 128, wp[4], bg, 0, 0, nullptr, 0, hWn, 64);
  k_gemm_wmma< 64, false, true><<<G(64), wv, 0, stream>>>(
      actB, 128, wp[3], bg, 0, 0, nullptr, 0, hWr, 64);
  // edge segment-max, residual combine
  k_edge_max<<<nb((long long)E * 64, TB), TB, 0, stream>>>(esrc, edst, cluster, hWn, agg, E);
  k_feat<<<nb((long long)MM * 64, TB), TB, 0, stream>>>(hWr, agg, bg, actA);
  // head MLP
  k_gemm_wmma< 64, true, false><<<G(112), wv, 0, stream>>>(
      actA, 128, wp[5], Q1b, 100, 1, actB, 128, nullptr, 0);
  k_gemm_wmma<128, true, false><<<G(112), wv, 0, stream>>>(
      actB, 128, wp[6], Q2b, 100, 1, actA, 128, nullptr, 0);
  k_gemm_wmma<128, true, false><<<G(112), wv, 0, stream>>>(
      actA, 128, wp[7], Q3b, 100, 1, actB, 128, nullptr, 0);
  k_gemm_wmma<128, false, true><<<G(32), wv, 0, stream>>>(
      actB, 128, wp[8], Q4b, 18, 0, nullptr, 0, qout, 32);
  // render
  k_render<<<nb(NPTS, TB), TB, 0, stream>>>(coords, cluster, qout, cent, (float*)d_out);
}
